// Attention_41558103556551
// MI455X (gfx1250) — matrix operand order repaired and checked by reading, not compiled, not run on hardware
//
#include <hip/hip_runtime.h>
#include <stdint.h>

#define DMODEL 1024
#define HEADS  16
#define DHEAD  64
#define BATCH  2
#define SEQ    2048

typedef __attribute__((ext_vector_type(16))) __bf16 v16bf;
typedef __attribute__((ext_vector_type(8)))  __bf16 v8bf;
typedef __attribute__((ext_vector_type(8)))  float  v8f;

// ---------- WMMA helpers ----------------------------------------------------

static __device__ __forceinline__ v8f wmma_bf16(v16bf a, v16bf b, v8f c) {
  // 8 args: (neg_a, A, neg_b, B, c_mod, C, reuse_a, reuse_b)
  return __builtin_amdgcn_wmma_f32_16x16x32_bf16(false, a, false, b,
                                                 (short)0, c, false, false);
}

// A-matrix fragment (16x32 bf16, M x K), row-major source with leading dim ldm.
// ISA layout: lanes 0-15 -> M=lane, K = 0..7 (V0..3) and 16..23 (V4..7);
//             lanes 16-31 -> M=lane-16, K = 8..15 and 24..31.
static __device__ __forceinline__ v16bf load_frag_a(const __bf16* base, int ldm, int lane) {
  const int r  = lane & 15;
  const int kb = (lane >> 4) << 3;           // 0 or 8
  const v8bf lo = *(const v8bf*)(base + (size_t)r * ldm + kb);
  const v8bf hi = *(const v8bf*)(base + (size_t)r * ldm + kb + 16);
  v16bf out;
#pragma unroll
  for (int i = 0; i < 8; ++i) { out[i] = lo[i]; out[i + 8] = hi[i]; }
  return out;
}

// B-matrix fragment (32x16 bf16, K x N). Source laid out N-major (we read row
// r = N of a [N][K] buffer): lanes 0-15 -> N=lane, K=0..7 and 16..23;
// lanes 16-31 -> N=lane-16, K=8..15 and 24..31.
static __device__ __forceinline__ v16bf load_frag_b(const __bf16* base, int ldm, int lane) {
  const int r  = lane & 15;
  const int kb = (lane >> 4) << 3;           // 0 or 8
  const v8bf lo = *(const v8bf*)(base + (size_t)r * ldm + kb);
  const v8bf hi = *(const v8bf*)(base + (size_t)r * ldm + kb + 16);
  v16bf out;
#pragma unroll
  for (int i = 0; i < 8; ++i) { out[i] = lo[i]; out[i + 8] = hi[i]; }
  return out;
}

// ---------- elementwise convert --------------------------------------------

__global__ void __launch_bounds__(256)
f32_to_bf16(const float* __restrict__ in, __bf16* __restrict__ out, int n) {
  int i = blockIdx.x * 256 + threadIdx.x;
  if (i < n) out[i] = (__bf16)in[i];
}

// ---------- generic bf16 GEMM: out = (A@B + bias) * scale -------------------
// A: [M,K] bf16 row-major, B: [K,N] bf16 row-major. Block = 128 thr (4 waves),
// tile 128(M) x 64(N): each wave owns 32 rows (2 A frags, 8 wmma / k-step).
// A fragments straight from global; B tile staged transposed in LDS.

template <bool OUT_F32>
__global__ void __launch_bounds__(128)
gemm_bf16(const __bf16* __restrict__ A, const __bf16* __restrict__ B,
          const float* __restrict__ bias, void* __restrict__ out,
          float scale, int M, int N, int K) {
  __shared__ __attribute__((aligned(16))) __bf16 ldsBt[64 * 40];

  const int tid   = threadIdx.x;
  const int lane  = tid & 31;
  const int wv    = tid >> 5;
  const int nbase = blockIdx.x * 64;
  const int mrow  = blockIdx.y * 128 + wv * 32;

  v8f zf = {};
  v8f acc[2][4];
#pragma unroll
  for (int s = 0; s < 2; ++s)
#pragma unroll
    for (int j = 0; j < 4; ++j) acc[s][j] = zf;

  const __bf16* Abase0 = A + (size_t)mrow * K;
  const __bf16* Abase1 = A + (size_t)(mrow + 16) * K;

  for (int kt = 0; kt < K; kt += 32) {
    __syncthreads();
    // stage B^T tile: ldsBt[n_local][k_local], padded stride 40
    for (int idx = tid; idx < 1024; idx += 128) {
      const int k = idx >> 5;
      const int n = (idx & 31) << 1;
      union { uint32_t u; __bf16 h[2]; } pk;
      pk.u = *(const uint32_t*)(B + (size_t)(kt + k) * N + nbase + n);
      ldsBt[(n)     * 40 + k] = pk.h[0];
      ldsBt[(n + 1) * 40 + k] = pk.h[1];
    }
    __syncthreads();

    const v16bf af0 = load_frag_a(Abase0 + kt, K, lane);
    const v16bf af1 = load_frag_a(Abase1 + kt, K, lane);
#pragma unroll
    for (int j = 0; j < 4; ++j) {
      const v16bf bf = load_frag_b(&ldsBt[(j * 16) * 40], 40, lane);
      acc[0][j] = wmma_bf16(af0, bf, acc[0][j]);
      acc[1][j] = wmma_bf16(af1, bf, acc[1][j]);
    }
  }

  // epilogue: C layout -> lane 0-15: N=lane, M=vgpr; lane 16-31: M=vgpr+8
  const int col = lane & 15;
  const int rhi = (lane >> 4) << 3;
#pragma unroll
  for (int s = 0; s < 2; ++s) {
#pragma unroll
    for (int j = 0; j < 4; ++j) {
      const int n  = nbase + j * 16 + col;
      const float bb = bias ? bias[n] : 0.0f;
#pragma unroll
      for (int i = 0; i < 8; ++i) {
        const int m    = mrow + s * 16 + rhi + i;
        const float vv = (acc[s][j][i] + bb) * scale;
        if (OUT_F32) ((float*)out)[(size_t)m * N + n] = vv;
        else         ((__bf16*)out)[(size_t)m * N + n] = (__bf16)vv;
      }
    }
  }
}

// ---------- fused flash attention (fully transposed) ------------------------
// Block = (b, h, 64 query rows), 4 waves x 16 queries. Keys stream in 32-wide
// steps. We compute S^T = K @ Q^T so that in the WMMA C layout each *lane*
// owns one query column: the online-softmax max/sum need only in-register
// reductions plus a single shfl_xor(16). The output is accumulated as
// O^T = V^T @ P^T, so rescale / 1/l are per-lane scalars (no broadcasts) and
// the epilogue packs 8 consecutive dh values per global_store_b128.

__global__ void __launch_bounds__(128)
attn_fused(const __bf16* __restrict__ Q, const __bf16* __restrict__ Kc,
           const __bf16* __restrict__ V, __bf16* __restrict__ O) {
  __shared__ __attribute__((aligned(16))) __bf16 ldsVt[64 * 40];
  __shared__ __attribute__((aligned(16))) __bf16 ldsP[4][16 * 32];

  const int tid  = threadIdx.x;
  const int lane = tid & 31;
  const int wv   = tid >> 5;
  const int qt   = blockIdx.x;
  const int h    = blockIdx.y;
  const int b    = blockIdx.z;

  const int q   = lane & 15;            // query owned by this lane
  const int rhi = (lane >> 4) << 3;     // 0 or 8 (C-layout M offset)

  const size_t qrow0 = (size_t)(b * SEQ + qt * 64 + wv * 16);
  const __bf16* qbase = Q + qrow0 * DMODEL + h * DHEAD;

  // Q^T as B operand: lane = query, half h holds dh 8h..8h+7 and 16+8h..16+8h+7
  const v16bf qb0 = load_frag_b(qbase, DMODEL, lane);       // dh 0..31
  const v16bf qb1 = load_frag_b(qbase + 32, DMODEL, lane);  // dh 32..63

  v8f zf = {};
  v8f acc[4];                           // O^T: M=dh (4x16), N=query
#pragma unroll
  for (int j = 0; j < 4; ++j) acc[j] = zf;
  float mq = -3.0e38f, lq = 0.0f;       // per-lane (per-query) stats

  for (int kt = 0; kt < SEQ; kt += 32) {
    __syncthreads();
    // stage V tile transposed: ldsVt[dh][key_local], padded stride 40
    const size_t vbase = ((size_t)(b * SEQ + kt)) * DMODEL + h * DHEAD;
    for (int idx = tid; idx < 1024; idx += 128) {
      const int k = idx >> 5;
      const int d = (idx & 31) << 1;
      union { uint32_t u; __bf16 h2[2]; } pk;
      pk.u = *(const uint32_t*)(V + vbase + (size_t)k * DMODEL + d);
      ldsVt[(d)     * 40 + k] = pk.h2[0];
      ldsVt[(d + 1) * 40 + k] = pk.h2[1];
    }
    __syncthreads();

    // S^T = K @ Q^T : two 16-key groups, dh contracted in 2 wmma steps
    v8f st[2];
#pragma unroll
    for (int g = 0; g < 2; ++g) {
      const __bf16* kb = Kc + ((size_t)(b * SEQ + kt + g * 16)) * DMODEL + h * DHEAD;
      const v16bf kf0 = load_frag_a(kb, DMODEL, lane);        // dh 0..31
      const v16bf kf1 = load_frag_a(kb + 32, DMODEL, lane);   // dh 32..63
      v8f sz = zf;
      sz = wmma_bf16(kf0, qb0, sz);
      sz = wmma_bf16(kf1, qb1, sz);
      st[g] = sz;      // C: N=query=lane, M=key=g*16 + rhi + i
    }

    // online softmax: 16 keys in-register per lane + partner half via xor-16
    float t = st[0][0];
#pragma unroll
    for (int i = 1; i < 8; ++i) t = fmaxf(t, st[0][i]);
#pragma unroll
    for (int i = 0; i < 8; ++i) t = fmaxf(t, st[1][i]);
    t = fmaxf(t, __shfl_xor(t, 16, 32));

    const float mn   = fmaxf(mq, t);
    const float resc = __expf(mq - mn);
    mq = mn;

    v8bf pk0, pk1;
    float ps = 0.0f;
#pragma unroll
    for (int i = 0; i < 8; ++i) {
      const float p0 = __expf(st[0][i] - mn);
      const float p1 = __expf(st[1][i] - mn);
      ps += p0 + p1;
      pk0[i] = (__bf16)p0;
      pk1[i] = (__bf16)p1;
    }
    ps += __shfl_xor(ps, 16, 32);
    lq = lq * resc + ps;
#pragma unroll
    for (int j = 0; j < 4; ++j)
#pragma unroll
      for (int i = 0; i < 8; ++i) acc[j][i] *= resc;

    // P^T -> per-wave LDS as [query][key] (two packed b128 stores)
    __bf16* pp = &ldsP[wv][0];
    *(v8bf*)(pp + q * 32 + rhi)      = pk0;   // keys  0..15 half
    *(v8bf*)(pp + q * 32 + 16 + rhi) = pk1;   // keys 16..31 half
    asm volatile("s_wait_dscnt 0" ::: "memory");  // LDS store -> load (same wave)

    // O^T += V^T @ P^T : A = V^T tiles (lane=dh row), B = P^T (lane=query)
    const v16bf pf = load_frag_b(pp, 32, lane);
#pragma unroll
    for (int j = 0; j < 4; ++j) {
      const v16bf vf = load_frag_a(&ldsVt[(j * 16) * 40], 40, lane);
      acc[j] = wmma_bf16(vf, pf, acc[j]);
    }
  }

  // normalize and store O^T: lane's query column, 8 consecutive dh per store
  const float inv = 1.0f / lq;
  __bf16* obase = O + (qrow0 + q) * DMODEL + h * DHEAD;
#pragma unroll
  for (int j = 0; j < 4; ++j) {
    v8bf o;
#pragma unroll
    for (int i = 0; i < 8; ++i) o[i] = (__bf16)(acc[j][i] * inv);
    *(v8bf*)(obase + j * 16 + rhi) = o;
  }
}

// ---------- host launch ------------------------------------------------------

extern "C" void kernel_launch(void* const* d_in, const int* in_sizes, int n_in,
                              void* d_out, int out_size, void* d_ws, size_t ws_size,
                              hipStream_t stream) {
  (void)in_sizes; (void)n_in; (void)out_size; (void)ws_size;

  const float* x  = (const float*)d_in[0];
  const float* Wq = (const float*)d_in[1];
  const float* bq = (const float*)d_in[2];
  const float* Wk = (const float*)d_in[3];
  const float* bk = (const float*)d_in[4];
  const float* Wv = (const float*)d_in[5];
  const float* bv = (const float*)d_in[6];
  const float* Wd = (const float*)d_in[7];
  const float* bd = (const float*)d_in[8];
  float* out = (float*)d_out;

  const size_t TOK = (size_t)BATCH * SEQ;       // 4096 tokens
  const size_t TM  = TOK * DMODEL;              // 4M elems
  const size_t WM  = (size_t)DMODEL * DMODEL;   // 1M elems

  __bf16* ws  = (__bf16*)d_ws;                  // 48 MiB bf16 workspace
  __bf16* xh  = ws;
  __bf16* qh  = xh + TM;
  __bf16* kh  = qh + TM;
  __bf16* vh  = kh + TM;
  __bf16* ph  = vh + TM;
  __bf16* wqh = ph + TM;
  __bf16* wkh = wqh + WM;
  __bf16* wvh = wkh + WM;
  __bf16* wdh = wvh + WM;

  f32_to_bf16<<<(int)((TM + 255) / 256), 256, 0, stream>>>(x,  xh,  (int)TM);
  f32_to_bf16<<<(int)((WM + 255) / 256), 256, 0, stream>>>(Wq, wqh, (int)WM);
  f32_to_bf16<<<(int)((WM + 255) / 256), 256, 0, stream>>>(Wk, wkh, (int)WM);
  f32_to_bf16<<<(int)((WM + 255) / 256), 256, 0, stream>>>(Wv, wvh, (int)WM);
  f32_to_bf16<<<(int)((WM + 255) / 256), 256, 0, stream>>>(Wd, wdh, (int)WM);

  dim3 gg(DMODEL / 64, (unsigned)(TOK / 128));  // (16, 32)
  dim3 gb(128);
  // softmax scale 1/sqrt(64) folded into Q
  gemm_bf16<false><<<gg, gb, 0, stream>>>(xh, wqh, bq, qh, 0.125f, (int)TOK, DMODEL, DMODEL);
  gemm_bf16<false><<<gg, gb, 0, stream>>>(xh, wkh, bk, kh, 1.0f,   (int)TOK, DMODEL, DMODEL);
  gemm_bf16<false><<<gg, gb, 0, stream>>>(xh, wvh, bv, vh, 1.0f,   (int)TOK, DMODEL, DMODEL);

  attn_fused<<<dim3(SEQ / 64, HEADS, BATCH), 128, 0, stream>>>(qh, kh, vh, ph);

  gemm_bf16<true><<<gg, gb, 0, stream>>>(ph, wdh, bd, (void*)out, 1.0f, (int)TOK, DMODEL, DMODEL);
}
